// GATLayer_82626580840514
// MI455X (gfx1250) — matrix operand order repaired and checked by reading, not compiled, not run on hardware
//
#include <hip/hip_runtime.h>
#include <hip/hip_bf16.h>

#define N_NODES 50000
#define IN_FEAT 512
#define OUTF    256     // HEADS * OUT_FEAT
#define MTILE   128     // rows of x per block (8 waves x 16)
#define KTILE   32      // one WMMA K-step per LDS stage

typedef __attribute__((ext_vector_type(16))) __bf16 v16bf;
typedef __attribute__((ext_vector_type(8)))  __bf16 bf16x8;
typedef __attribute__((ext_vector_type(8)))  float  v8f;

union Frag16 { v16bf v; bf16x8 h[2]; __bf16 e[16]; };

// ---- gfx1250 async copy: global -> LDS, tracked by ASYNCcnt ----
__device__ __forceinline__ void async_copy_b128(unsigned lds_off,
                                                unsigned long long gaddr)
{
    asm volatile("global_load_async_to_lds_b128 %0, %1, off"
                 :: "v"(lds_off), "v"(gaddr) : "memory");
}
__device__ __forceinline__ void async_wait0()
{
    asm volatile("s_wait_asynccnt 0x0" ::: "memory");
}
__device__ __forceinline__ unsigned lds_addr(const void* p)
{
    // generic shared-aperture address: low 32 bits == LDS byte offset
    return (unsigned)(unsigned long long)(unsigned long long)(uintptr_t)p;
}
__device__ __forceinline__ void split2(float f, __bf16& hi, __bf16& lo)
{
    hi = (__bf16)f;
    lo = (__bf16)(f - (float)hi);
}

// ---------------------------------------------------------------------------
// W prep: split fp32 W (256 x 512, row-major [col][k]) into bf16 hi/lo planes
// so the GEMM can stage W K-tiles with pure async DMA.
// ---------------------------------------------------------------------------
__global__ __launch_bounds__(256) void gat_prep_w(
    const float* __restrict__ W, __bf16* __restrict__ whi,
    __bf16* __restrict__ wlo, int n)
{
    int i = blockIdx.x * blockDim.x + threadIdx.x;
    if (i >= n) return;
    __bf16 hi, lo;
    split2(W[i], hi, lo);
    whi[i] = hi;
    wlo[i] = lo;
}

// ---------------------------------------------------------------------------
// GEMM: h[n, 256] = x[n, 512] @ W^T with bf16 hi/lo split (3 WMMAs -> near
// fp32 accuracy) on v_wmma_f32_16x16x32_bf16.  Double-buffered LDS staging
// via global_load_async_to_lds_b128 overlaps DMA with WMMA compute.
// Block: 256 threads = 8 wave32; wave w owns rows [blockM+16w, +16) for all
// 256 output columns (16 resident v8f accumulators).
// ---------------------------------------------------------------------------
__global__ __launch_bounds__(256) void gat_gemm_bf16x3(
    const float* __restrict__ x, const __bf16* __restrict__ whi,
    const __bf16* __restrict__ wlo, float* __restrict__ h)
{
    __shared__ __align__(16) float  xbuf [2][MTILE * KTILE];  // fp32 x tile
    __shared__ __align__(16) __bf16 whib [2][OUTF * KTILE];   // W hi tile
    __shared__ __align__(16) __bf16 wlob [2][OUTF * KTILE];   // W lo tile

    const int tid   = threadIdx.x;
    const int wv    = tid >> 5;       // wave id 0..7 (wave32)
    const int lane  = tid & 31;
    const int half  = lane >> 4;      // lane half selects K sub-range
    const int l16   = lane & 15;
    const int mBase = blockIdx.x * MTILE;

    // per-thread staging geometry:
    //   x tile : thread t copies 64 B  = 16 floats of row (t>>1), k-half (t&1)
    //   W tiles: thread t copies 64 B  = 32 bf16 of column t
    const int xrow  = tid >> 1;
    const int xk    = (tid & 1) * 16;
    const int gxrow = mBase + xrow;

    auto stage = [&](int buf, int k0) {
        // ---- x tile (raw fp32; bf16 split happens at fragment build) ----
        if (gxrow < N_NODES) {
            unsigned long long g =
                (unsigned long long)(uintptr_t)&x[(size_t)gxrow * IN_FEAT + k0 + xk];
            unsigned l = lds_addr(&xbuf[buf][xrow * KTILE + xk]);
            #pragma unroll
            for (int c = 0; c < 4; ++c)
                async_copy_b128(l + c * 16, g + c * 16);
        } else {
            float4 z = {0.f, 0.f, 0.f, 0.f};
            float4* p = (float4*)&xbuf[buf][xrow * KTILE + xk];
            #pragma unroll
            for (int c = 0; c < 4; ++c) p[c] = z;
        }
        // ---- W hi/lo tiles (pre-split planes, pure DMA) ----
        {
            unsigned long long gh =
                (unsigned long long)(uintptr_t)&whi[(size_t)tid * IN_FEAT + k0];
            unsigned long long gl =
                (unsigned long long)(uintptr_t)&wlo[(size_t)tid * IN_FEAT + k0];
            unsigned lh = lds_addr(&whib[buf][tid * KTILE]);
            unsigned ll = lds_addr(&wlob[buf][tid * KTILE]);
            #pragma unroll
            for (int c = 0; c < 4; ++c) {
                async_copy_b128(lh + c * 16, gh + c * 16);
                async_copy_b128(ll + c * 16, gl + c * 16);
            }
        }
    };

    const v8f vzero = {0.f, 0.f, 0.f, 0.f, 0.f, 0.f, 0.f, 0.f};
    v8f acc[16];
    #pragma unroll
    for (int t = 0; t < 16; ++t) acc[t] = vzero;

    int cur = 0;
    stage(cur, 0);
    async_wait0();
    __syncthreads();

    for (int k0 = 0; k0 < IN_FEAT; k0 += KTILE) {
        const int nxt = cur ^ 1;
        if (k0 + KTILE < IN_FEAT) stage(nxt, k0 + KTILE);  // overlap DMA

        // ---- A fragments: read fp32 from LDS, split to bf16 hi/lo.
        // ISA 16-bit A 16x32 layout: lanes 0-15 hold K {0..7, 16..23},
        // lanes 16-31 hold K {8..15, 24..31}. ----
        Frag16 ah, al;
        {
            const float* xr = &xbuf[cur][(wv * 16 + l16) * KTILE];
            const int kb = half * 8;
            #pragma unroll
            for (int j = 0; j < 8; ++j) {
                __bf16 h0, l0, h1, l1;
                split2(xr[kb + j],      h0, l0);
                split2(xr[kb + 16 + j], h1, l1);
                ah.e[j] = h0;     al.e[j] = l0;
                ah.e[8 + j] = h1; al.e[8 + j] = l1;
            }
        }

        // ---- 16 n-tiles; B layout: col = lane%16, lane half h holds K
        // {8h..8h+7, 16+8h..16+8h+7}, matching A ----
        #pragma unroll
        for (int t = 0; t < 16; ++t) {
            Frag16 bh, bl;
            const int col  = t * 16 + l16;
            const int koff = half * 8;
            bh.h[0] = *reinterpret_cast<const bf16x8*>(&whib[cur][col * KTILE + koff]);
            bh.h[1] = *reinterpret_cast<const bf16x8*>(&whib[cur][col * KTILE + koff + 16]);
            bl.h[0] = *reinterpret_cast<const bf16x8*>(&wlob[cur][col * KTILE + koff]);
            bl.h[1] = *reinterpret_cast<const bf16x8*>(&wlob[cur][col * KTILE + koff + 16]);

            acc[t] = __builtin_amdgcn_wmma_f32_16x16x32_bf16(
                false, ah.v, false, bh.v, (short)0, acc[t], false, false);
            acc[t] = __builtin_amdgcn_wmma_f32_16x16x32_bf16(
                false, ah.v, false, bl.v, (short)0, acc[t], false, false);
            acc[t] = __builtin_amdgcn_wmma_f32_16x16x32_bf16(
                false, al.v, false, bh.v, (short)0, acc[t], false, false);
        }

        async_wait0();      // next tile's DMA done (this wave's ASYNCcnt)
        __syncthreads();    // everyone done reading cur + DMA visible
        cur = nxt;
    }

    // ---- writeback: C/D layout: VGPR r -> M = r (lanes 0-15) / r+8
    // (lanes 16-31), N = lane%16 ----
    #pragma unroll
    for (int t = 0; t < 16; ++t) {
        const int col = t * 16 + l16;
        #pragma unroll
        for (int r = 0; r < 8; ++r) {
            const int row = mBase + wv * 16 + half * 8 + r;
            if (row < N_NODES) h[(size_t)row * OUTF + col] = acc[t][r];
        }
    }
}

// ---------------------------------------------------------------------------
// Scatter-sum: one wave32 per edge; h and sum are L2-resident (51 MB each in
// the 192 MB L2), so f32 atomics run at L2 rate.
// ---------------------------------------------------------------------------
__global__ __launch_bounds__(256) void gat_scatter(
    const int* __restrict__ ei, const float* __restrict__ h,
    float* __restrict__ sum, float* __restrict__ cnt, int nE)
{
    long gid = (long)blockIdx.x * blockDim.x + threadIdx.x;
    int e    = (int)(gid >> 5);
    int lane = (int)(gid & 31);
    if (e >= nE) return;
    int s = ei[e];        // src
    int d = ei[nE + e];   // dst
    const float* hs = h   + (size_t)s * OUTF;
    float*       sd = sum + (size_t)d * OUTF;
    #pragma unroll
    for (int i = 0; i < OUTF / 32; ++i)
        atomicAdd(&sd[lane + i * 32], hs[lane + i * 32]);
    if (lane == 0) atomicAdd(&cnt[d], 1.0f);
}

// out currently holds h; overwrite with mean where cnt > 0
__global__ __launch_bounds__(256) void gat_finalize(
    float* __restrict__ out, const float* __restrict__ sum,
    const float* __restrict__ cnt, int total)
{
    int i = blockIdx.x * blockDim.x + threadIdx.x;
    if (i >= total) return;
    float c = cnt[i >> 8];  // OUTF == 256
    if (c > 0.f) out[i] = sum[i] / c;
}

__global__ __launch_bounds__(256) void gat_zero(float* __restrict__ p, int n)
{
    int i = blockIdx.x * blockDim.x + threadIdx.x;
    if (i < n) p[i] = 0.f;
}

extern "C" void kernel_launch(void* const* d_in, const int* in_sizes, int n_in,
                              void* d_out, int out_size, void* d_ws, size_t ws_size,
                              hipStream_t stream)
{
    const float* x  = (const float*)d_in[0];
    const float* W  = (const float*)d_in[1];
    const int*   ei = (const int*)d_in[2];
    float* out = (float*)d_out;

    // ws layout (all 16B aligned):
    //   sum : 50000*256 f32   (51,200,000 B)
    //   cnt : 50000     f32   (   200,000 B)
    //   whi : 256*512   bf16  (   262,144 B)
    //   wlo : 256*512   bf16  (   262,144 B)
    float*  sum = (float*)d_ws;
    float*  cnt = sum + (size_t)N_NODES * OUTF;
    __bf16* whi = (__bf16*)(cnt + N_NODES);
    __bf16* wlo = whi + (size_t)OUTF * IN_FEAT;
    const int nE = in_sizes[2] / 2;

    // zero scratch every call (harness does not re-poison between replays)
    const int zn = N_NODES * OUTF + N_NODES;
    gat_zero<<<(zn + 255) / 256, 256, 0, stream>>>(sum, zn);

    // split W into bf16 hi/lo planes (pure-DMA staging in the GEMM)
    const int wn = OUTF * IN_FEAT;
    gat_prep_w<<<(wn + 255) / 256, 256, 0, stream>>>(W, whi, wlo, wn);

    // h -> d_out (needed as fallback for nodes with no in-edges)
    gat_gemm_bf16x3<<<(N_NODES + MTILE - 1) / MTILE, 256, 0, stream>>>(x, whi, wlo, out);

    long sthreads = (long)nE * 32;
    gat_scatter<<<(int)((sthreads + 255) / 256), 256, 0, stream>>>(ei, out, sum, cnt, nE);

    const int total = N_NODES * OUTF;
    gat_finalize<<<(total + 255) / 256, 256, 0, stream>>>(out, sum, cnt, total);
}
